// MetricClassifier_53584011985262
// MI455X (gfx1250) — compile-verified, matrix operand order checked by reading, not run on hardware
//
#include <hip/hip_runtime.h>

#define NUM_CLASSES 512
#define D_FEAT      1024
#define N_CTX       65536
#define N_TGT       65536
#define SCALE       50.0f
#define EPS         1e-30f

typedef __attribute__((ext_vector_type(8)))  float  v8f;
typedef __attribute__((ext_vector_type(16))) __bf16 v16bf;

// ---------------------------------------------------------------------------
// Zero scratch (sums + counts)
// ---------------------------------------------------------------------------
__global__ void mc_zero_kernel(float* __restrict__ p, int n) {
    int i = blockIdx.x * blockDim.x + threadIdx.x;
    if (i < n) p[i] = 0.0f;
}

// ---------------------------------------------------------------------------
// Per-class counts (float, matches reference segment_sum of ones)
// ---------------------------------------------------------------------------
__global__ void mc_count_kernel(const int* __restrict__ labels,
                                float* __restrict__ counts) {
    int i = blockIdx.x * blockDim.x + threadIdx.x;  // 0..N_CTX-1
    atomicAdd(&counts[labels[i]], 1.0f);
}

// ---------------------------------------------------------------------------
// Per-class feature sums via global f32 atomics (2 MB sums -> L2 resident)
// ---------------------------------------------------------------------------
__global__ void mc_sum_kernel(const float* __restrict__ ctx,
                              const int* __restrict__ labels,
                              float* __restrict__ sums) {
    size_t i = (size_t)blockIdx.x * blockDim.x + threadIdx.x;  // 0..N_CTX*D-1
    int row = (int)(i >> 10);
    int col = (int)(i & (D_FEAT - 1));
    atomicAdd(&sums[(size_t)labels[row] * D_FEAT + col], ctx[i]);
}

// ---------------------------------------------------------------------------
// means = sums / max(count,1); write f32 means to d_out, and pack bf16 hi/lo
// B-fragments in the exact WMMA 16x16x32 bf16 B layout:
//   fragment index = ((ntile*32 + kchunk)*32 + lane)*16 + e
//   lane = half*16 + (class & 15);  k&31: half=(kk>>3)&1, e=(kk&7)|((kk&16)>>1)
// ---------------------------------------------------------------------------
__global__ void mc_finalize_means_kernel(const float* __restrict__ sums,
                                         const float* __restrict__ counts,
                                         float* __restrict__ means_out,
                                         __bf16* __restrict__ Bhi,
                                         __bf16* __restrict__ Blo) {
    int i = blockIdx.x * blockDim.x + threadIdx.x;  // 0..NUM_CLASSES*D-1
    int c = i >> 10;
    int k = i & (D_FEAT - 1);
    float m = sums[i] / fmaxf(counts[c], 1.0f);
    means_out[i] = m;

    __bf16 hi = (__bf16)m;
    __bf16 lo = (__bf16)(m - (float)hi);

    int nt = c >> 4;
    int kc = k >> 5;
    int kk = k & 31;
    int half = (kk >> 3) & 1;
    int e    = (kk & 7) | ((kk & 16) >> 1);
    int lane = half * 16 + (c & 15);
    size_t fi = (((size_t)nt * 32 + kc) * 32 + lane) * 16 + e;
    Bhi[fi] = hi;
    Blo[fi] = lo;
}

// ---------------------------------------------------------------------------
// Row L2 norms: one wave32 per row, float4 loads, shuffle reduction
// ---------------------------------------------------------------------------
__global__ void mc_row_norm_kernel(const float* __restrict__ X,
                                   float* __restrict__ norms, int nrows) {
    int row  = blockIdx.x * (blockDim.x >> 5) + (threadIdx.x >> 5);
    int lane = threadIdx.x & 31;
    if (row >= nrows) return;
    const float* xp = X + (size_t)row * D_FEAT;
    float s = 0.0f;
    #pragma unroll
    for (int k = lane * 4; k < D_FEAT; k += 32 * 4) {
        float4 v = *(const float4*)(xp + k);
        s += v.x * v.x + v.y * v.y + v.z * v.z + v.w * v.w;
    }
    #pragma unroll
    for (int off = 16; off > 0; off >>= 1) s += __shfl_down(s, off, 32);
    if (lane == 0) norms[row] = fmaxf(sqrtf(s), EPS);
}

// ---------------------------------------------------------------------------
// Cosine-similarity GEMM: logits = 50 * (A @ B^T) / (tnorm mnorm)
// fp32 emulated by bf16x3 WMMA: AhiBhi + AhiBlo + AloBhi (16x16x32 bf16).
// Block = 8 waves x 16 rows = 128 M rows. B fragments are shared by all 8
// waves, so they are staged L2->LDS with gfx1250 async copies
// (global_load_async_to_lds_b128, ASYNCcnt) in 64 KB stages:
//   stage = 4 N-tiles x 8 K-chunks, hi (32 KB) + lo (32 KB).
// ---------------------------------------------------------------------------
__global__ __launch_bounds__(256)
void mc_cosine_gemm_kernel(const float* __restrict__ A,
                           const __bf16* __restrict__ Bhi,
                           const __bf16* __restrict__ Blo,
                           const float* __restrict__ t_norm,
                           const float* __restrict__ m_norm,
                           float* __restrict__ out) {
    __shared__ __align__(16) unsigned char smem[65536];

    const int tid  = threadIdx.x;
    const int lane = tid & 31;
    const int wave = tid >> 5;
    const int half = lane >> 4;   // K-half selector for A/B fragments
    const int row  = lane & 15;   // M row (A) / N col (B,C) within tile
    const int m0   = blockIdx.x * 128 + wave * 16;

    const float* arow = A + (size_t)(m0 + row) * D_FEAT;
    const v16bf* lds_hi = (const v16bf*)smem;            // [4*8*32] fragments
    const v16bf* lds_lo = (const v16bf*)(smem + 32768);

    // 50 / ||t|| for the 8 C-rows this lane owns (M = m0 + r + 8*half)
    float inv_tn[8];
    #pragma unroll
    for (int r = 0; r < 8; ++r)
        inv_tn[r] = SCALE / t_norm[m0 + r + 8 * half];

    for (int ng = 0; ng < 8; ++ng) {
        v8f acc0 = 0, acc1 = 0, acc2 = 0, acc3 = 0;

        for (int kg = 0; kg < 4; ++kg) {       // K stages of 8 chunks (K=256)
            // ---- stage B fragments L2 -> LDS via async copy ----
            __syncthreads();                   // prior stage fully consumed
            #pragma unroll
            for (int j = 0; j < 16; ++j) {
                int idx = tid + j * 256;       // 4096 B128 transfers total
                int hl  = idx >> 11;           // 0 = hi, 1 = lo
                int sub = idx & 2047;
                int t   = sub >> 9;            // tile within N-group
                int off = sub & 511;           // B128 within 8 KB region
                const __bf16* g = (hl ? Blo : Bhi)
                    + (((size_t)(ng * 4 + t) * 32 + kg * 8) * 32) * 16
                    + (size_t)off * 8;
                unsigned lds_off = (unsigned)(uintptr_t)(smem + hl * 32768
                                                       + t * 8192 + off * 16);
                asm volatile("global_load_async_to_lds_b128 %0, %1, off"
                             :: "v"(lds_off), "v"(g) : "memory");
            }
            asm volatile("s_wait_asynccnt 0x0" ::: "memory");
            __syncthreads();

            // ---- compute on staged fragments ----
            for (int kcl = 0; kcl < 8; ++kcl) {
                const int kc = kg * 8 + kcl;
                // A fragment: lane holds row (lane&15), Ks = kc*32+half*8+{0..7}
                // and kc*32+16+half*8+{0..7}   (ISA 16-bit A 16x32 layout)
                const int kb = kc * 32 + half * 8;
                float4 f0 = *(const float4*)(arow + kb);
                float4 f1 = *(const float4*)(arow + kb + 4);
                float4 f2 = *(const float4*)(arow + kb + 16);
                float4 f3 = *(const float4*)(arow + kb + 20);
                float af[16] = { f0.x, f0.y, f0.z, f0.w, f1.x, f1.y, f1.z, f1.w,
                                 f2.x, f2.y, f2.z, f2.w, f3.x, f3.y, f3.z, f3.w };
                v16bf ahi, alo;
                #pragma unroll
                for (int e = 0; e < 16; ++e) {
                    __bf16 h = (__bf16)af[e];
                    ahi[e] = h;
                    alo[e] = (__bf16)(af[e] - (float)h);
                }

                #pragma unroll
                for (int t = 0; t < 4; ++t) {
                    const int fi = (t * 8 + kcl) * 32 + lane;
                    v16bf bhi = lds_hi[fi];
                    v16bf blo = lds_lo[fi];
                    v8f a = (t == 0) ? acc0 : (t == 1) ? acc1
                          : (t == 2) ? acc2 : acc3;
                    a = __builtin_amdgcn_wmma_f32_16x16x32_bf16(
                            false, ahi, false, bhi, (short)0, a, false, false);
                    a = __builtin_amdgcn_wmma_f32_16x16x32_bf16(
                            false, ahi, false, blo, (short)0, a, false, false);
                    a = __builtin_amdgcn_wmma_f32_16x16x32_bf16(
                            false, alo, false, bhi, (short)0, a, false, false);
                    if (t == 0) acc0 = a; else if (t == 1) acc1 = a;
                    else if (t == 2) acc2 = a; else acc3 = a;
                }
            }
        }

        // Epilogue: C/D layout lane l vgpr r -> M = m0 + r + 8*half,
        //           N = nt*16 + (l&15)
        #pragma unroll
        for (int t = 0; t < 4; ++t) {
            const int nt = ng * 4 + t;
            const int n  = nt * 16 + row;
            const float invmn = 1.0f / m_norm[n];
            const v8f a = (t == 0) ? acc0 : (t == 1) ? acc1
                        : (t == 2) ? acc2 : acc3;
            #pragma unroll
            for (int r = 0; r < 8; ++r) {
                const int m = m0 + r + 8 * half;
                out[(size_t)m * NUM_CLASSES + n] = a[r] * inv_tn[r] * invmn;
            }
        }
    }
}

// ---------------------------------------------------------------------------
// Launch
// ---------------------------------------------------------------------------
extern "C" void kernel_launch(void* const* d_in, const int* in_sizes, int n_in,
                              void* d_out, int out_size, void* d_ws, size_t ws_size,
                              hipStream_t stream) {
    const float* ctx    = (const float*)d_in[0];
    const int*   labels = (const int*)d_in[1];
    const float* tgt    = (const float*)d_in[2];

    float* logits    = (float*)d_out;                                   // [N_TGT, C]
    float* means_out = logits + (size_t)N_TGT * NUM_CLASSES;            // [C, D]

    // Workspace layout (~4.5 MB)
    float*  sums   = (float*)d_ws;                        // C*D
    float*  counts = sums + (size_t)NUM_CLASSES * D_FEAT; // C
    float*  t_norm = counts + NUM_CLASSES;                // N_TGT
    float*  m_norm = t_norm + N_TGT;                      // C
    __bf16* Bhi    = (__bf16*)(m_norm + NUM_CLASSES);     // C*D bf16 (frag layout)
    __bf16* Blo    = Bhi + (size_t)NUM_CLASSES * D_FEAT;

    const int nz = NUM_CLASSES * D_FEAT + NUM_CLASSES;    // sums+counts contiguous
    mc_zero_kernel<<<(nz + 255) / 256, 256, 0, stream>>>(sums, nz);

    mc_count_kernel<<<N_CTX / 256, 256, 0, stream>>>(labels, counts);

    mc_sum_kernel<<<(unsigned)(((size_t)N_CTX * D_FEAT) / 256), 256, 0, stream>>>(
        ctx, labels, sums);

    mc_finalize_means_kernel<<<(NUM_CLASSES * D_FEAT) / 256, 256, 0, stream>>>(
        sums, counts, means_out, Bhi, Blo);

    mc_row_norm_kernel<<<NUM_CLASSES / 8, 256, 0, stream>>>(means_out, m_norm, NUM_CLASSES);
    mc_row_norm_kernel<<<N_TGT / 8, 256, 0, stream>>>(tgt, t_norm, N_TGT);

    mc_cosine_gemm_kernel<<<N_TGT / 128, 256, 0, stream>>>(
        tgt, Bhi, Blo, t_norm, m_norm, logits);
}
